// RWKV6Mixin_5188320493605
// MI455X (gfx1250) — matrix operand order repaired and checked by reading, not compiled, not run on hardware
//
#include <hip/hip_runtime.h>
#include <hip/hip_bf16.h>

// RWKV6Attention forward for MI455X (gfx1250, wave32, WMMA).
// v3: GEMM templated on epilogue (no runtime branches) and sub-tile guards
// removed (all dims are multiples of 32 in this net), so the inner loop is
// pure b128 loads + v_wmma_f32_16x16x32_bf16 with EXEC untouched (ISA 7.12
// requires EXEC all-ones for WMMA). Operands pre-packed to bf16 once.

#define DEVINL __device__ __forceinline__

typedef __attribute__((ext_vector_type(16))) __bf16 v16bf;
typedef __attribute__((ext_vector_type(8)))  __bf16 bf16x8;
typedef __attribute__((ext_vector_type(8)))  float  v8f;

namespace cfg {
constexpr int H    = 2048;
constexpr int NH   = 16;
constexpr int KD   = 1024;
constexpr int VD   = 1024;
constexpr int DK   = 64;
constexpr int DV   = 64;
constexpr int RLOW = 64;
constexpr int GLOW = 64;
constexpr int Bc   = 2;
constexpr int Tc   = 1024;
constexpr int MT   = Bc * Tc;      // 2048 tokens
constexpr float EPS = 1e-5f;
}

enum { EP_NONE = 0, EP_TANH = 1, EP_LERP = 2, EP_DECAY = 3, EP_BIAS = 4 };

DEVINL bf16x8 ld8(const __bf16* p) {
  return *reinterpret_cast<const bf16x8*>(p);
}

DEVINL v16bf combine16(bf16x8 lo, bf16x8 hi) {
  v16bf r;
#pragma unroll
  for (int i = 0; i < 8; ++i) { r[i] = lo[i]; r[8 + i] = hi[i]; }
  return r;
}

// A fragment (16-bit A 16x32, ISA 7.12.2): lane<16 -> row m0+lane, K in
// {0..7} u {16..23}; lane>=16 -> +8.  Two contiguous 16B chunks per lane.
DEVINL v16bf load_a(const __bf16* __restrict__ A, long lda,
                    int m0, int kk, int lane) {
  const long row = m0 + (lane & 15);
  const int  kb  = (lane >> 4) ? 8 : 0;
  const __bf16* p = A + row * lda + kk + kb;
  return combine16(ld8(p), ld8(p + 16));
}

// B fragment (16-bit B 32x16): lane<16 -> col n0+lane, K in {0..7} u
// {16..23}; lane>=16 -> +8.  With Bt[n*ldb + k] two 16B chunks per lane.
DEVINL v16bf load_b(const __bf16* __restrict__ Bt, long ldb,
                    int kk, int n0, int lane) {
  const long col = n0 + (lane & 15);
  const int  kb  = (lane >> 4) ? 8 : 0;
  const __bf16* p = Bt + col * ldb + kk + kb;
  return combine16(ld8(p), ld8(p + 16));
}

// C store + fused epilogue (compile-time specialized).
// C/D layout: VGPR i -> row m0 + i + (lane>=16?8:0), col = n0 + (lane&15).
template <int EP, bool OUT_BF16>
DEVINL void store_c(const v8f& c, void* __restrict__ C, long ldc,
                    int m0, int n0, int lane,
                    const float* __restrict__ bias,
                    const float* __restrict__ ex,
                    const float* __restrict__ ed) {
  const int col   = n0 + (lane & 15);
  const int rbase = m0 + ((lane >> 4) << 3);
  float bb = 0.0f;
  if (EP == EP_LERP || EP == EP_DECAY || EP == EP_BIAS) bb = bias[col];
#pragma unroll
  for (int i = 0; i < 8; ++i) {
    const long row = rbase + i;
    float f = c[i];
    if (EP == EP_TANH) {
      f = tanhf(f);
    } else if (EP == EP_LERP) {
      const long idx = row * ldc + col;
      f = ex[idx] + ed[idx] * (f + bb);
    } else if (EP == EP_DECAY) {
      f = expf(-expf(f + bb));
    } else if (EP == EP_BIAS) {
      f = f + bb;
    }
    if (OUT_BF16)
      ((__bf16*)C)[row * ldc + col] = (__bf16)f;
    else
      ((float*)C)[row * ldc + col] = f;
  }
}

// ---------------------------------------------------------------------------
// bf16 WMMA GEMM: C[M,N] = epilogue(A[M,K] @ Bt[N,K]^T).
// Requires M % 32 == 0, N % 32 == 0, K % 32 == 0 (true for this network).
// Block = 256 threads = 8 waves (2x4); each wave owns a full 32x32 tile.
// Inner loop: 8 x global_load_b128 + 4 x v_wmma, no EXEC manipulation.
// ---------------------------------------------------------------------------
template <int EP, bool OUT_BF16>
__global__ __launch_bounds__(256) void k_gemm(
    const __bf16* __restrict__ A, long lda,
    const __bf16* __restrict__ Bt, long ldb,
    void* __restrict__ C, long ldc,
    int M, int N, int K,
    const float* __restrict__ bias,
    const float* __restrict__ ex, const float* __restrict__ ed) {
  const int lane = threadIdx.x & 31;
  const int wave = threadIdx.x >> 5;
  const int mBase = blockIdx.y * 64 + (wave >> 2) * 32;
  const int nBase = blockIdx.x * 128 + (wave & 3) * 32;
  if (mBase >= M || nBase >= N) return;   // wave-uniform; EXEC stays all-ones

  v8f c00 = {}, c01 = {}, c10 = {}, c11 = {};
  for (int kk = 0; kk < K; kk += 32) {
    const v16bf a0 = load_a(A, lda, mBase, kk, lane);
    const v16bf a1 = load_a(A, lda, mBase + 16, kk, lane);
    const v16bf b0 = load_b(Bt, ldb, kk, nBase, lane);
    const v16bf b1 = load_b(Bt, ldb, kk, nBase + 16, lane);
    c00 = __builtin_amdgcn_wmma_f32_16x16x32_bf16(false, a0, false, b0,
                                                  (short)0, c00, false, false);
    c01 = __builtin_amdgcn_wmma_f32_16x16x32_bf16(false, a0, false, b1,
                                                  (short)0, c01, false, false);
    c10 = __builtin_amdgcn_wmma_f32_16x16x32_bf16(false, a1, false, b0,
                                                  (short)0, c10, false, false);
    c11 = __builtin_amdgcn_wmma_f32_16x16x32_bf16(false, a1, false, b1,
                                                  (short)0, c11, false, false);
  }

  store_c<EP, OUT_BF16>(c00, C, ldc, mBase, nBase, lane, bias, ex, ed);
  store_c<EP, OUT_BF16>(c01, C, ldc, mBase, nBase + 16, lane, bias, ex, ed);
  store_c<EP, OUT_BF16>(c10, C, ldc, mBase + 16, nBase, lane, bias, ex, ed);
  store_c<EP, OUT_BF16>(c11, C, ldc, mBase + 16, nBase + 16, lane, bias, ex, ed);
}

// ---------------------------------------------------------------------------
// Weight transpose + convert: W[k*N + n] (fp32) -> Wt[n*K + k] (bf16).
// ---------------------------------------------------------------------------
__global__ __launch_bounds__(256) void k_transpose_cvt(
    const float* __restrict__ W, int K, int N, __bf16* __restrict__ Wt) {
  __shared__ float tile[32][33];
  const int kb = blockIdx.y * 32, nb = blockIdx.x * 32;
  const int tx = threadIdx.x & 31, ty = threadIdx.x >> 5;  // 32 x 8
#pragma unroll
  for (int i = ty; i < 32; i += 8) {
    const int k = kb + i, n = nb + tx;
    tile[i][tx] = (k < K && n < N) ? W[(long)k * N + n] : 0.0f;
  }
  __syncthreads();
#pragma unroll
  for (int i = ty; i < 32; i += 8) {
    const int n = nb + i, k = kb + tx;
    if (n < N && k < K) Wt[(long)n * K + k] = (__bf16)tile[tx][i];
  }
}

// Straight fp32 -> bf16 convert (for Wx2, already [N,K]-shaped per slice).
__global__ __launch_bounds__(256) void k_cvt(const float* __restrict__ in,
                                             __bf16* __restrict__ out, long n) {
  const long i = (long)blockIdx.x * blockDim.x + threadIdx.x;
  if (i < n) out[i] = (__bf16)in[i];
}

// ---------------------------------------------------------------------------
// prep: delta = time_shift(x) - x (fp32) ; z0 = x + delta*mu_x (bf16)
// ---------------------------------------------------------------------------
__global__ __launch_bounds__(256) void k_prep(const float* __restrict__ x,
                                              const float* __restrict__ mu_x,
                                              float* __restrict__ delta,
                                              __bf16* __restrict__ z0, long n) {
  const long i = (long)blockIdx.x * blockDim.x + threadIdx.x;
  if (i >= n) return;
  const long h   = i % cfg::H;
  const long tok = i / cfg::H;
  const long t   = tok % cfg::Tc;
  const float xi   = x[i];
  const float prev = (t == 0) ? 0.0f : x[i - cfg::H];
  const float d    = prev - xi;
  delta[i] = d;
  z0[i]    = (__bf16)(xi + d * mu_x[h]);
}

// ---------------------------------------------------------------------------
// Sequential scan: one block per (b, head); 64x64 state in registers.
//   o_t = r_t . (S + u * k_t v_t^T) ;  S = diag(decay_t) S + k_t v_t^T
// ---------------------------------------------------------------------------
__global__ __launch_bounds__(256) void k_recurrence(
    const float* __restrict__ r, const float* __restrict__ k,
    const float* __restrict__ v, const float* __restrict__ decay,
    const float* __restrict__ bonus, float* __restrict__ o) {
  const int bh = blockIdx.x;
  const int b  = bh / cfg::NH;
  const int h  = bh % cfg::NH;
  const int tid = threadIdx.x;
  const int dv  = tid & 63;
  const int g4  = tid >> 6;

  float S[16];
#pragma unroll
  for (int i = 0; i < 16; ++i) S[i] = 0.0f;
  float u[16];
#pragma unroll
  for (int i = 0; i < 16; ++i) u[i] = bonus[h * cfg::DK + g4 * 16 + i];

  __shared__ float sh_r[64], sh_k[64], sh_v[64], sh_d[64];
  __shared__ float sh_o[4][64];

  for (int t = 0; t < cfg::Tc; ++t) {
    const long base = ((long)b * cfg::Tc + t) * (long)cfg::KD + (long)h * 64;
    if (tid < 64) {
      sh_r[tid] = r[base + tid];
      sh_k[tid] = k[base + tid];
    } else if (tid < 128) {
      const int j = tid - 64;
      sh_v[j] = v[base + j];
      sh_d[j] = decay[base + j];
    }
    __syncthreads();

    const float vt = sh_v[dv];
    float acc = 0.0f;
#pragma unroll
    for (int i = 0; i < 16; ++i) {
      const int dk = g4 * 16 + i;
      const float kv = sh_k[dk] * vt;
      acc += sh_r[dk] * (S[i] + u[i] * kv);
      S[i] = sh_d[dk] * S[i] + kv;
    }
    sh_o[g4][dv] = acc;
    __syncthreads();
    if (g4 == 0)
      o[base + dv] = sh_o[0][dv] + sh_o[1][dv] + sh_o[2][dv] + sh_o[3][dv];
    __syncthreads();
  }
}

// ---------------------------------------------------------------------------
// Per-(token,head) layernorm + swish gate; writes bf16 for the final GEMM.
// ---------------------------------------------------------------------------
__global__ __launch_bounds__(256) void k_ln_gate(
    const float* __restrict__ o, const float* __restrict__ g,
    const float* __restrict__ ln_w, const float* __restrict__ ln_b,
    __bf16* __restrict__ og, long rows) {
  const long row = (long)blockIdx.x * 4 + (threadIdx.x >> 6);
  const int  l   = threadIdx.x & 63;
  const int  rg  = threadIdx.x >> 6;
  __shared__ float s_sum[4][64];
  __shared__ float s_sq[4][64];

  float val = 0.0f, gv = 0.0f;
  if (row < rows) {
    val = o[row * 64 + l];
    gv  = g[row * 64 + l];
  }
  s_sum[rg][l] = val;
  s_sq[rg][l]  = val * val;
  __syncthreads();
#pragma unroll
  for (int off = 32; off > 0; off >>= 1) {
    if (l < off) {
      s_sum[rg][l] += s_sum[rg][l + off];
      s_sq[rg][l]  += s_sq[rg][l + off];
    }
    __syncthreads();
  }
  if (row < rows) {
    const float mean = s_sum[rg][0] * (1.0f / 64.0f);
    const float var  = s_sq[rg][0] * (1.0f / 64.0f) - mean * mean;
    const float on   = (val - mean) * rsqrtf(var + cfg::EPS) * ln_w[l] + ln_b[l];
    const float sig  = 1.0f / (1.0f + expf(-gv));
    og[row * 64 + l] = (__bf16)(on * gv * sig);
  }
}

// ---------------------------------------------------------------------------
// Host side
// ---------------------------------------------------------------------------
template <int EP, bool OUT_BF16>
static inline void gemm(hipStream_t s,
                        const __bf16* A, long lda,
                        const __bf16* Bt, long ldb,
                        void* C, long ldc, int M, int N, int K,
                        const float* bias = nullptr,
                        const float* ex = nullptr, const float* ed = nullptr) {
  dim3 grid((unsigned)((N + 127) / 128), (unsigned)((M + 63) / 64));
  k_gemm<EP, OUT_BF16><<<grid, 256, 0, s>>>(A, lda, Bt, ldb, C, ldc, M, N, K,
                                            bias, ex, ed);
}

static inline void transpose_cvt(hipStream_t s, const float* W, int K, int N,
                                 __bf16* Wt) {
  dim3 grid((unsigned)((N + 31) / 32), (unsigned)((K + 31) / 32));
  k_transpose_cvt<<<grid, 256, 0, s>>>(W, K, N, Wt);
}

extern "C" void kernel_launch(void* const* d_in, const int* in_sizes, int n_in,
                              void* d_out, int out_size, void* d_ws, size_t ws_size,
                              hipStream_t stream) {
  (void)in_sizes; (void)n_in; (void)out_size; (void)ws_size;
  using namespace cfg;

  const float* x      = (const float*)d_in[0];
  const float* mu_x   = (const float*)d_in[1];
  const float* Wx1    = (const float*)d_in[2];   // [H, 5*RLOW]
  const float* Wx2    = (const float*)d_in[3];   // [H, 5, RLOW]
  const float* x_bias = (const float*)d_in[4];   // [5, H]
  const float* Wr     = (const float*)d_in[5];   // [H, KD]
  const float* Wk     = (const float*)d_in[6];
  const float* Wv     = (const float*)d_in[7];
  const float* Ww_a   = (const float*)d_in[8];   // [H, GLOW]
  const float* Ww_b   = (const float*)d_in[9];   // [GLOW, KD]
  const float* bw     = (const float*)d_in[10];
  const float* Wg_a   = (const float*)d_in[11];
  const float* Wg_b   = (const float*)d_in[12];  // [GLOW, VD]
  const float* bg     = (const float*)d_in[13];
  const float* bonus  = (const float*)d_in[14];  // [NH, DK]
  const float* ln_w   = (const float*)d_in[15];
  const float* ln_b   = (const float*)d_in[16];
  const float* Wo     = (const float*)d_in[17];  // [VD, H]
  float* out = (float*)d_out;

  // Byte-based workspace allocator, 256B-aligned chunks.
  char* wsb = (char*)d_ws;
  size_t off = 0;
  auto alloc = [&](size_t bytes) {
    void* p = wsb + off;
    off = (off + bytes + 255) & ~(size_t)255;
    return p;
  };
  // fp32 buffers
  float* delta = (float*)alloc((size_t)MT * H * 4);
  float* rb    = (float*)alloc((size_t)MT * KD * 4);
  float* kb    = (float*)alloc((size_t)MT * KD * 4);
  float* vb    = (float*)alloc((size_t)MT * VD * 4);
  float* db    = (float*)alloc((size_t)MT * KD * 4);
  float* gb    = (float*)alloc((size_t)MT * VD * 4);
  float* ob    = (float*)alloc((size_t)MT * VD * 4);
  // bf16 activation buffers
  __bf16* z0   = (__bf16*)alloc((size_t)MT * H * 2);
  __bf16* lrb  = (__bf16*)alloc((size_t)MT * 5 * RLOW * 2);
  __bf16* xmb  = (__bf16*)alloc((size_t)MT * H * 2);       // reused per stream
  __bf16* wlrb = (__bf16*)alloc((size_t)MT * GLOW * 2);
  __bf16* glrb = (__bf16*)alloc((size_t)MT * GLOW * 2);
  __bf16* ogb  = (__bf16*)alloc((size_t)MT * VD * 2);
  // bf16 weights, pre-transposed to [N, K]
  __bf16* Wx1t = (__bf16*)alloc((size_t)5 * RLOW * H * 2);
  __bf16* Wx2b = (__bf16*)alloc((size_t)H * 5 * RLOW * 2); // straight convert
  __bf16* Wrt  = (__bf16*)alloc((size_t)KD * H * 2);
  __bf16* Wkt  = (__bf16*)alloc((size_t)KD * H * 2);
  __bf16* Wvt  = (__bf16*)alloc((size_t)VD * H * 2);
  __bf16* Wwat = (__bf16*)alloc((size_t)GLOW * H * 2);
  __bf16* Wwbt = (__bf16*)alloc((size_t)KD * GLOW * 2);
  __bf16* Wgat = (__bf16*)alloc((size_t)GLOW * H * 2);
  __bf16* Wgbt = (__bf16*)alloc((size_t)VD * GLOW * 2);
  __bf16* Wot  = (__bf16*)alloc((size_t)H * VD * 2);

  // 0. pack weights (cheap: ~10M elements total)
  transpose_cvt(stream, Wx1, H, 5 * RLOW, Wx1t);
  transpose_cvt(stream, Wr,  H, KD, Wrt);
  transpose_cvt(stream, Wk,  H, KD, Wkt);
  transpose_cvt(stream, Wv,  H, VD, Wvt);
  transpose_cvt(stream, Ww_a, H, GLOW, Wwat);
  transpose_cvt(stream, Ww_b, GLOW, KD, Wwbt);
  transpose_cvt(stream, Wg_a, H, GLOW, Wgat);
  transpose_cvt(stream, Wg_b, GLOW, VD, Wgbt);
  transpose_cvt(stream, Wo,  VD, H, Wot);
  {
    const long n = (long)H * 5 * RLOW;
    k_cvt<<<(unsigned)((n + 255) / 256), 256, 0, stream>>>(Wx2, Wx2b, n);
  }

  // 1. time-shift delta (fp32) and z0 = x + delta*mu_x (bf16)
  {
    const long n = (long)MT * H;
    k_prep<<<(unsigned)((n + 255) / 256), 256, 0, stream>>>(x, mu_x, delta, z0, n);
  }

  // 2. lr = tanh(z0 @ Wx1)  -> bf16 [MT, 5*RLOW]
  gemm<EP_TANH, true>(stream, z0, H, Wx1t, H, lrb, 5 * RLOW, MT, 5 * RLOW, H);

  // 3. per stream s: xm = x + delta*(lr_s @ Wx2_s^T + x_bias_s) (bf16),
  //    then the stream's projection GEMM(s).
  for (int s = 0; s < 5; ++s) {
    gemm<EP_LERP, true>(stream, lrb + s * RLOW, 5 * RLOW,
                        Wx2b + s * RLOW, 5 * RLOW,
                        xmb, H, MT, H, RLOW,
                        x_bias + (size_t)s * H, x, delta);
    switch (s) {
      case 0:  // r = xm @ Wr
        gemm<EP_NONE, false>(stream, xmb, H, Wrt, H, rb, KD, MT, KD, H);
        break;
      case 1:  // decay = exp(-exp(tanh(xm @ Ww_a) @ Ww_b + bw))
        gemm<EP_TANH, true>(stream, xmb, H, Wwat, H, wlrb, GLOW, MT, GLOW, H);
        gemm<EP_DECAY, false>(stream, wlrb, GLOW, Wwbt, GLOW, db, KD,
                              MT, KD, GLOW, bw);
        break;
      case 2:  // k = xm @ Wk
        gemm<EP_NONE, false>(stream, xmb, H, Wkt, H, kb, KD, MT, KD, H);
        break;
      case 3:  // v = xm @ Wv
        gemm<EP_NONE, false>(stream, xmb, H, Wvt, H, vb, VD, MT, VD, H);
        break;
      case 4:  // g = tanh(xm @ Wg_a) @ Wg_b + bg
        gemm<EP_TANH, true>(stream, xmb, H, Wgat, H, glrb, GLOW, MT, GLOW, H);
        gemm<EP_BIAS, false>(stream, glrb, GLOW, Wgbt, GLOW, gb, VD,
                             MT, VD, GLOW, bg);
        break;
    }
  }

  // 4. linear-attention scan
  k_recurrence<<<Bc * NH, 256, 0, stream>>>(rb, kb, vb, db, bonus, ob);

  // 5. per-head layernorm + swish gate -> bf16
  {
    const long rows = (long)MT * NH;
    k_ln_gate<<<(unsigned)((rows + 3) / 4), 256, 0, stream>>>(ob, gb, ln_w, ln_b,
                                                              ogb, rows);
  }

  // 6. out = og @ Wo (fp32 out)
  gemm<EP_NONE, false>(stream, ogb, VD, Wot, VD, out, H, MT, H, VD);
}
